// MultiHeadAttention_62440234549404
// MI455X (gfx1250) — matrix operand order repaired and checked by reading, compiled, not run on hardware
//
#include <hip/hip_runtime.h>
#include <hip/hip_bf16.h>

typedef __bf16 bf16;
typedef __attribute__((ext_vector_type(16))) __bf16 v16bf;
typedef __attribute__((ext_vector_type(8)))  __bf16 v8bf;
typedef __attribute__((ext_vector_type(8)))  float  v8f;

#define B_SZ 2
#define S_LEN 2048
#define D_MODEL 1024
#define N_HEADS 16
#define HD 64
#define M_ROWS (B_SZ * S_LEN)   /* 4096 */
#define QKV_N  (3 * D_MODEL)    /* 3072 */
#define NEG_INF_F (-9.0e15f)
#define LDS_PAD 8               /* rows padded to 72 halfs = 144B: conflict-free */

__device__ __forceinline__ v8f wmma_bf16f32(v16bf a, v16bf b, v8f c) {
  // D = A(16x32 bf16) * B(32x16 bf16) + C(16x16 f32)
  return __builtin_amdgcn_wmma_f32_16x16x32_bf16(false, a, false, b, (short)0, c,
                                                 false, false);
}

// A fragment: 16x32 bf16 tile from row-major [m, k] source (leading dim lda).
// ISA layout: lane(0-15)=M, hi half-wave holds K+8/K+24 groups. Two 16B loads.
__device__ __forceinline__ v16bf load_a_frag(const bf16* A, int lda, int m0,
                                             int k0, int lane) {
  const int m  = m0 + (lane & 15);
  const int hi = lane >> 4;
  const bf16* p = A + (size_t)m * lda + k0 + hi * 8;
  v8bf lo = *(const v8bf*)(p);        // K = base .. base+7
  v8bf hh = *(const v8bf*)(p + 16);   // K = base+16 .. base+23
  v16bf r;
#pragma unroll
  for (int i = 0; i < 8; ++i) { r[i] = lo[i]; r[i + 8] = hh[i]; }
  return r;
}

// B fragment: 32x16 bf16 tile where B(k,n) = W[n][k], W row-major [n, k].
// lane(0-15)=N; hi half-wave holds K+8/K+24 groups (same K order as A fragment).
__device__ __forceinline__ v16bf load_b_frag(const bf16* W, int ldw, int n0,
                                             int k0, int lane) {
  const int n  = n0 + (lane & 15);
  const int hi = lane >> 4;
  const bf16* p = W + (size_t)n * ldw + k0 + hi * 8;
  v8bf lo = *(const v8bf*)(p);
  v8bf hh = *(const v8bf*)(p + 16);
  v16bf r;
#pragma unroll
  for (int i = 0; i < 8; ++i) { r[i] = lo[i]; r[i + 8] = hh[i]; }
  return r;
}

__global__ __launch_bounds__(256) void cvt_f32_to_bf16(const float* __restrict__ in,
                                                       bf16* __restrict__ out, int n) {
  int i = blockIdx.x * 256 + threadIdx.x;
  if (i < n) out[i] = (bf16)in[i];
}

// C[M,N] = A[M,K](bf16) @ W[N,K]^T(bf16) + bias[N], 4 waves/block, 64x64/wave.
template <bool OUT_BF16>
__global__ __launch_bounds__(128) void gemm_bias_wmma(
    const bf16* __restrict__ A, const bf16* __restrict__ W,
    const float* __restrict__ bias, void* __restrict__ Cout,
    int M, int N, int K) {
  const int lane = threadIdx.x & 31;
  const int wave = threadIdx.x >> 5;
  const int m0 = blockIdx.x * 128 + (wave >> 1) * 64;
  const int n0 = blockIdx.y * 128 + (wave & 1) * 64;
  const v8f vzero = {};
  v8f acc[4][4];
#pragma unroll
  for (int i = 0; i < 4; ++i)
#pragma unroll
    for (int j = 0; j < 4; ++j) acc[i][j] = vzero;

  for (int k0 = 0; k0 < K; k0 += 32) {
    v16bf af[4], bfm[4];
#pragma unroll
    for (int i = 0; i < 4; ++i) af[i] = load_a_frag(A, K, m0 + 16 * i, k0, lane);
#pragma unroll
    for (int j = 0; j < 4; ++j) bfm[j] = load_b_frag(W, K, n0 + 16 * j, k0, lane);
#pragma unroll
    for (int i = 0; i < 4; ++i)
#pragma unroll
      for (int j = 0; j < 4; ++j) acc[i][j] = wmma_bf16f32(af[i], bfm[j], acc[i][j]);
  }

  const int hi = lane >> 4, nn = lane & 15;
#pragma unroll
  for (int j = 0; j < 4; ++j) {
    const int col = n0 + j * 16 + nn;
    const float bv = bias[col];
#pragma unroll
    for (int i = 0; i < 4; ++i) {
#pragma unroll
      for (int r = 0; r < 8; ++r) {
        const int row = m0 + i * 16 + hi * 8 + r;
        const float v = acc[i][j][r] + bv;
        if (OUT_BF16) ((bf16*)Cout)[(size_t)row * N + col] = (bf16)v;
        else          ((float*)Cout)[(size_t)row * N + col] = v;
      }
    }
  }
}

// Flash attention: grid (S/128, B*H), 128 threads (4 waves). Wave owns 32 q rows.
// qkv row layout (3072 cols): per head h: [q(64) | k(64) | v(64)] at col h*192.
// K tile is staged once per block via async global->LDS copies (ASYNCcnt path);
// V tile is transposed into LDS; probabilities bounce through LDS as bf16.
__global__ __launch_bounds__(128) void attn_flash_wmma(
    const bf16* __restrict__ qkv, const int* __restrict__ mask,
    bf16* __restrict__ ctx) {
  __shared__ bf16 Pt[128][HD + LDS_PAD];   // P tile (bf16 probs), stride 72
  __shared__ bf16 Kt[64][64 + LDS_PAD];    // K tile row-major [kv][d], stride 72
  __shared__ bf16 Vt[HD][64 + LDS_PAD];    // V tile transposed [d][kv], stride 72

  const int lane = threadIdx.x & 31;
  const int wave = threadIdx.x >> 5;
  const int hi = lane >> 4, nn = lane & 15;
  const int b = blockIdx.y >> 4;
  const int h = blockIdx.y & 15;
  const int q0 = blockIdx.x * 128;   // query offset within S
  const int wq = wave * 32;          // wave's sub-offset within chunk
  const int ldq = QKV_N;
  const bf16* Qp = qkv + (size_t)b * S_LEN * ldq + h * (3 * HD);
  const bf16* Kp = Qp + HD;
  const bf16* Vp = Qp + 2 * HD;
  const float scale = 0.125f;        // 1/sqrt(64)

  // Async-copy setup: K tile = 64 rows x 128B; 512 16B chunks over 128 threads.
  // chunk id = tid + i*128 : row = id>>3, chunk-in-row = id&7.
  const unsigned kt_base = (unsigned)(size_t)&Kt[0][0];  // LDS offset of Kt

  // Q fragments resident in registers: 2 M-frags x 2 K-steps
  v16bf qf[2][2];
#pragma unroll
  for (int mi = 0; mi < 2; ++mi)
#pragma unroll
    for (int ks = 0; ks < 2; ++ks)
      qf[mi][ks] = load_a_frag(Qp, ldq, q0 + wq + mi * 16, ks * 32, lane);

  const v8f vzero = {};
  v8f acc_o[2][4];
#pragma unroll
  for (int mi = 0; mi < 2; ++mi)
#pragma unroll
    for (int oj = 0; oj < 4; ++oj) acc_o[mi][oj] = vzero;
  float m_run[2][8], l_run[2][8];
#pragma unroll
  for (int mi = 0; mi < 2; ++mi)
#pragma unroll
    for (int r = 0; r < 8; ++r) { m_run[mi][r] = -3.0e38f; l_run[mi][r] = 0.0f; }

  for (int kv0 = 0; kv0 < S_LEN; kv0 += 64) {
    __syncthreads();  // previous iteration's readers of Kt/Vt are done

    // --- async stage K tile: global -> LDS, no VGPR round trip ---
#pragma unroll
    for (int i = 0; i < 4; ++i) {
      const int id  = threadIdx.x + i * 128;
      const int row = id >> 3;          // kv row in tile
      const int ch  = id & 7;           // 16B chunk within 128B row
      const unsigned ldsoff = kt_base + (unsigned)(row * (64 + LDS_PAD) * 2 + ch * 16);
      const bf16* gp = Kp + (size_t)(kv0 + row) * ldq + ch * 8;
      asm volatile("global_load_async_to_lds_b128 %0, %1, off"
                   :: "v"(ldsoff), "v"(gp)
                   : "memory");
    }

    // --- stage V^T (overlaps with the async K copy) ---
    {
      const int kvr = threadIdx.x >> 1;
      const int dbase = (threadIdx.x & 1) * 32;
      const bf16* vp = Vp + (size_t)(kv0 + kvr) * ldq + dbase;
#pragma unroll
      for (int dd = 0; dd < 32; dd += 8) {
        v8bf t = *(const v8bf*)(vp + dd);
#pragma unroll
        for (int j = 0; j < 8; ++j) Vt[dbase + dd + j][kvr] = t[j];
      }
    }

    // hint next tile's mask rows into cache while we compute
    if (kv0 + 64 < S_LEN) {
      const int qrow_pf = q0 + wq + hi * 8;
      __builtin_prefetch(&mask[(size_t)qrow_pf * S_LEN + kv0 + 64 + nn * 4], 0, 1);
    }

    asm volatile("s_wait_asynccnt 0" ::: "memory");  // K tile landed in LDS
    __syncthreads();                                 // Kt + Vt visible to all waves

    // logits S = Q @ K^T for this 128x64 slab (B frags from LDS Kt)
    v8f sacc[2][4];
#pragma unroll
    for (int mi = 0; mi < 2; ++mi)
#pragma unroll
      for (int nj = 0; nj < 4; ++nj) sacc[mi][nj] = vzero;
#pragma unroll
    for (int nj = 0; nj < 4; ++nj) {
#pragma unroll
      for (int ks = 0; ks < 2; ++ks) {
        v16bf kf = load_b_frag(&Kt[0][0], 64 + LDS_PAD, nj * 16, ks * 32, lane);
#pragma unroll
        for (int mi = 0; mi < 2; ++mi)
          sacc[mi][nj] = wmma_bf16f32(qf[mi][ks], kf, sacc[mi][nj]);
      }
    }

    // scale + mask + online softmax; write bf16 probs to LDS
#pragma unroll
    for (int mi = 0; mi < 2; ++mi) {
#pragma unroll
      for (int r = 0; r < 8; ++r) {
        const int qrow = q0 + wq + mi * 16 + hi * 8 + r;
        float sv[4];
        float mx = -3.0e38f;
#pragma unroll
        for (int nj = 0; nj < 4; ++nj) {
          const int kcol = kv0 + nj * 16 + nn;
          float s = sacc[mi][nj][r] * scale;
          s = (mask[(size_t)qrow * S_LEN + kcol] == 0) ? NEG_INF_F : s;
          sv[nj] = s;
          mx = fmaxf(mx, s);
        }
        // row reduce across the 16-lane N group (stays within half-wave)
        mx = fmaxf(mx, __shfl_xor(mx, 1, 32));
        mx = fmaxf(mx, __shfl_xor(mx, 2, 32));
        mx = fmaxf(mx, __shfl_xor(mx, 4, 32));
        mx = fmaxf(mx, __shfl_xor(mx, 8, 32));
        const float mnew  = fmaxf(m_run[mi][r], mx);
        const float alpha = __expf(m_run[mi][r] - mnew);
        m_run[mi][r] = mnew;
        float ps = 0.0f;
#pragma unroll
        for (int nj = 0; nj < 4; ++nj) {
          const float p = __expf(sv[nj] - mnew);
          ps += p;
          Pt[wq + mi * 16 + hi * 8 + r][nj * 16 + nn] = (bf16)p;
        }
        ps += __shfl_xor(ps, 1, 32);
        ps += __shfl_xor(ps, 2, 32);
        ps += __shfl_xor(ps, 4, 32);
        ps += __shfl_xor(ps, 8, 32);
        l_run[mi][r] = l_run[mi][r] * alpha + ps;
#pragma unroll
        for (int oj = 0; oj < 4; ++oj) acc_o[mi][oj][r] *= alpha;
      }
    }
    __syncthreads();  // Pt visible

    // O += P @ V  (A frags from Pt, B frags from transposed Vt)
#pragma unroll
    for (int ks = 0; ks < 2; ++ks) {
      v16bf pa[2];
#pragma unroll
      for (int mi = 0; mi < 2; ++mi)
        pa[mi] = load_a_frag(&Pt[0][0], HD + LDS_PAD, wq + mi * 16, ks * 32, lane);
#pragma unroll
      for (int oj = 0; oj < 4; ++oj) {
        v16bf vb = load_b_frag(&Vt[0][0], 64 + LDS_PAD, oj * 16, ks * 32, lane);
#pragma unroll
        for (int mi = 0; mi < 2; ++mi)
          acc_o[mi][oj] = wmma_bf16f32(pa[mi], vb, acc_o[mi][oj]);
      }
    }
  }

  // normalize and write ctx[b*S + s][h*64 + d] as bf16
#pragma unroll
  for (int mi = 0; mi < 2; ++mi) {
#pragma unroll
    for (int r = 0; r < 8; ++r) {
      const float inv = 1.0f / l_run[mi][r];
      const size_t row = (size_t)b * S_LEN + q0 + wq + mi * 16 + hi * 8 + r;
#pragma unroll
      for (int oj = 0; oj < 4; ++oj) {
        const int col = h * HD + oj * 16 + nn;
        ctx[row * D_MODEL + col] = (bf16)(acc_o[mi][oj][r] * inv);
      }
    }
  }
}

extern "C" void kernel_launch(void* const* d_in, const int* in_sizes, int n_in,
                              void* d_out, int out_size, void* d_ws, size_t ws_size,
                              hipStream_t stream) {
  (void)in_sizes; (void)n_in; (void)out_size; (void)ws_size;
  const float* x     = (const float*)d_in[0];
  const int*   mask  = (const int*)  d_in[1];
  const float* w_qkv = (const float*)d_in[2];
  const float* b_qkv = (const float*)d_in[3];
  const float* w_o   = (const float*)d_in[4];
  const float* b_o   = (const float*)d_in[5];
  float* out = (float*)d_out;
  char* ws = (char*)d_ws;

  // workspace layout (bytes)
  bf16* x_bf   = (bf16*)(ws + 0);                  //  8,388,608
  bf16* wq_bf  = (bf16*)(ws + (size_t)8388608);    //  6,291,456
  bf16* wo_bf  = (bf16*)(ws + (size_t)14680064);   //  2,097,152
  bf16* qkv_bf = (bf16*)(ws + (size_t)16777216);   // 25,165,824
  bf16* ctx_bf = (bf16*)(ws + (size_t)41943040);   //  8,388,608 -> 50,331,648 total

  cvt_f32_to_bf16<<<(M_ROWS * D_MODEL + 255) / 256, 256, 0, stream>>>(
      x, x_bf, M_ROWS * D_MODEL);
  cvt_f32_to_bf16<<<(QKV_N * D_MODEL + 255) / 256, 256, 0, stream>>>(
      w_qkv, wq_bf, QKV_N * D_MODEL);
  cvt_f32_to_bf16<<<(D_MODEL * D_MODEL + 255) / 256, 256, 0, stream>>>(
      w_o, wo_bf, D_MODEL * D_MODEL);

  // qkv = x @ w_qkv^T + b_qkv   (M=4096, N=3072, K=1024), bf16 output
  gemm_bias_wmma<true><<<dim3(M_ROWS / 128, QKV_N / 128), 128, 0, stream>>>(
      x_bf, wq_bf, b_qkv, (void*)qkv_bf, M_ROWS, QKV_N, D_MODEL);

  // per-(b,h) flash attention over 128-row query chunks
  attn_flash_wmma<<<dim3(S_LEN / 128, B_SZ * N_HEADS), 128, 0, stream>>>(
      qkv_bf, mask, ctx_bf);

  // out = ctx @ w_o^T + b_o     (M=4096, N=1024, K=1024), f32 output
  gemm_bias_wmma<false><<<dim3(M_ROWS / 128, D_MODEL / 128), 128, 0, stream>>>(
      ctx_bf, wo_bf, b_o, (void*)out, M_ROWS, D_MODEL, D_MODEL);
}
